// Backflow_noMF_10634339024911
// MI455X (gfx1250) — matrix operand order repaired and checked by reading, not compiled, not run on hardware
//
#include <hip/hip_runtime.h>

typedef __attribute__((ext_vector_type(16))) _Float16 v16h;
typedef __attribute__((ext_vector_type(8)))  float    v8f;
typedef __attribute__((ext_vector_type(4)))  float    v4f;

#define BATCH      65536
#define N_ORB      64
#define NF         32          // 16 up + 16 dn
#define KDIM       128         // 2*N_ORB
#define NCOLS      2048        // N_ORB*NF
#define TILE_B     16          // batches per workgroup
#define AROWS      17          // 16 real slots + 1 trash slot (branchless scatter)

// ---------------------------------------------------------------------------
// Kernel 0: W (f32, [128][2048]) -> WhT (f16, [2048][128])  (B-operand layout:
// column-major so each WMMA B fragment is one contiguous 32B load per lane)
// ---------------------------------------------------------------------------
__global__ __launch_bounds__(256) void prep_w_kernel(const float* __restrict__ W,
                                                     _Float16* __restrict__ WhT) {
    int idx = blockIdx.x * 256 + threadIdx.x;      // 0 .. 2048*128-1
    int col = idx >> 7;                            // F column (0..2047)
    int k   = idx & 127;                           // orbital (0..127)
    WhT[idx] = (_Float16)W[(size_t)k * NCOLS + col];
}

// ---------------------------------------------------------------------------
// Kernel 1: per 16-batch tile:
//   phase 0: load n, build f16 A-operand staging + occupied-row inverse maps
//   phase 1: WMMA GEMM F_tile = n_tile @ W, branchless scatter of needed rows
//   phase 2: two 16x16 LU decompositions per batch (half-wave each), logdet
// ---------------------------------------------------------------------------
__global__ __launch_bounds__(256) void backflow_kernel(const int* __restrict__ n,
                                                       const _Float16* __restrict__ WhT,
                                                       float* __restrict__ out) {
    __shared__ _Float16    nh[TILE_B * KDIM];              // 4 KB f16 staging of n
    __shared__ signed char inv2[2 * N_ORB * TILE_B];       // 2 KB [half][r][m] -> slot|-1
    __shared__ float       A_lds[TILE_B * 2 * AROWS * 16]; // ~34 KB gathered A matrices

    const int tid   = threadIdx.x;
    const int lane  = tid & 31;
    const int wv    = tid >> 5;           // wave id 0..7
    const int mrow  = lane & 15;
    const int hiSel = lane >> 4;          // 0: lanes 0-15, 1: lanes 16-31
    const int tile  = blockIdx.x;
    const int* nrow = n + (size_t)tile * TILE_B * KDIM;

    // ---------------- phase 0 ----------------
    #pragma unroll
    for (int q = 0; q < 8; ++q) {                          // 2048 elements of n -> f16
        int idx = q * 256 + tid;
        nh[idx] = (_Float16)(float)nrow[idx];
    }
    #pragma unroll
    for (int q = 0; q < 8; ++q) inv2[q * 256 + tid] = -1;  // 2048 bytes
    __syncthreads();
    if (tid < 32) {                                        // one thread per (batch, spin)
        int m = tid >> 1, s = tid & 1;
        int cnt = 0;
        for (int o = 0; o < N_ORB; ++o) {
            int occ = nrow[m * KDIM + s * N_ORB + o];
            if (occ && cnt < 16) {
                inv2[(s * N_ORB + o) * TILE_B + m] = (signed char)cnt;
                ++cnt;
            }
        }
    }
    __syncthreads();

    // ---------------- phase 1: WMMA GEMM ----------------
    // A fragments (contiguous k = 16*half + s layout, matching B), reused across all 16 column tiles
    v16h afrag[4];
    #pragma unroll
    for (int kk = 0; kk < 4; ++kk) {
        v16h a;
        #pragma unroll
        for (int h = 0; h < 8; ++h)
            a[h] = nh[mrow * KDIM + kk * 32 + hiSel * 16 + h];
        #pragma unroll
        for (int h = 0; h < 8; ++h)
            a[8 + h] = nh[mrow * KDIM + kk * 32 + hiSel * 16 + 8 + h];
        afrag[kk] = a;
    }

    for (int t = 0; t < 16; ++t) {
        const int ct = wv + 8 * t;                         // column tile 0..127
        const _Float16* bp = WhT + (size_t)(ct * 16 + mrow) * KDIM + hiSel * 16;
        if (t < 15)
            __builtin_prefetch(WhT + (size_t)((ct + 8) * 16 + mrow) * KDIM, 0, 0);

        v8f acc = {};
        #pragma unroll
        for (int kk = 0; kk < 4; ++kk) {
            v16h b = *(const v16h*)(bp + kk * 32);         // 32B-aligned contiguous
            acc = __builtin_amdgcn_wmma_f32_16x16x32_f16(
                false, afrag[kk], false, b, (short)0, acc, false, false);
        }

        // Branchless scatter: ct = 2*r + half ; F col = r*32 + half*16 + j.
        // One ds_load_b64 fetches slots for all 8 local batches this lane owns;
        // slot 16 is a trash row so every store is unconditional.
        const int r = ct >> 1, half = ct & 1;
        const long long packed =
            *(const long long*)&inv2[(half * N_ORB + r) * TILE_B + (hiSel << 3)];
        #pragma unroll
        for (int v = 0; v < 8; ++v) {
            int ii   = (int)(signed char)(packed >> (8 * v));
            int slot = (ii >= 0) ? ii : 16;
            int m    = v + (hiSel << 3);                   // C/D row = local batch
            A_lds[(m * 2 + half) * (AROWS * 16) + slot * 16 + mrow] = acc[v];
        }
    }
    __syncthreads();

    // ---------------- phase 2: LU with partial pivoting ----------------
    const float PI = 3.14159265358979323846f;
    const int base = hiSel << 4;
    const int i = mrow;                                    // my row index

    for (int mb = wv; mb < TILE_B; mb += 8) {
        const v4f* src = (const v4f*)&A_lds[(mb * 2 + hiSel) * (AROWS * 16) + i * 16];
        v4f r0 = src[0], r1 = src[1], r2 = src[2], r3 = src[3];
        float row[16] = { r0.x, r0.y, r0.z, r0.w, r1.x, r1.y, r1.z, r1.w,
                          r2.x, r2.y, r2.z, r2.w, r3.x, r3.y, r3.z, r3.w };
        float logabs = 0.0f;
        int neg = 0;

        #pragma unroll
        for (int k = 0; k < 16; ++k) {
            // argmax |row[k]| over rows >= k, within this 16-lane half
            float cand = (i >= k) ? fabsf(row[k]) : -1.0f;
            int cidx = i;
            #pragma unroll
            for (int off = 8; off >= 1; off >>= 1) {
                float oc = __shfl_xor(cand, off, 32);
                int   oi = __shfl_xor(cidx, off, 32);
                if (oc > cand || (oc == cand && oi < cidx)) { cand = oc; cidx = oi; }
            }
            const int p = cidx;
            // swap rows k <-> p (U part only)
            const int srcl = base + ((i == k) ? p : (i == p) ? k : i);
            #pragma unroll
            for (int c = k; c < 16; ++c) row[c] = __shfl(row[c], srcl, 32);
            // broadcast pivot row
            float pr[16];
            #pragma unroll
            for (int c = k; c < 16; ++c) pr[c] = __shfl(row[c], base + k, 32);
            const float dpiv = pr[k];
            neg ^= (p != k) ? 1 : 0;
            neg ^= (dpiv < 0.0f) ? 1 : 0;
            logabs += logf(fabsf(dpiv));
            // eliminate
            const float f = (i > k) ? row[k] / dpiv : 0.0f;
            #pragma unroll
            for (int c = k + 1; c < 16; ++c) row[c] = fmaf(-f, pr[c], row[c]);
        }

        const float la_u = __shfl(logabs, 0, 32);
        const float la_d = __shfl(logabs, 16, 32);
        const int   ng_u = __shfl(neg, 0, 32);
        const int   ng_d = __shfl(neg, 16, 32);
        if (lane == 0) {
            size_t gb = (size_t)tile * TILE_B + mb;
            out[2 * gb]     = la_u + la_d;                 // Re(logdet_u + logdet_d)
            out[2 * gb + 1] = PI * (float)(ng_u + ng_d);   // Im: pi per negative det
        }
    }
}

// ---------------------------------------------------------------------------
extern "C" void kernel_launch(void* const* d_in, const int* in_sizes, int n_in,
                              void* d_out, int out_size, void* d_ws, size_t ws_size,
                              hipStream_t stream) {
    const int*   nptr = (const int*)d_in[0];     // (B, 128) int32 occupancy
    const float* W    = (const float*)d_in[1];   // (128, 2048) f32
    float*       out  = (float*)d_out;           // complex64 as interleaved f32
    _Float16*    WhT  = (_Float16*)d_ws;         // 512 KB: W^T in f16

    prep_w_kernel<<<(NCOLS * KDIM) / 256, 256, 0, stream>>>(W, WhT);
    backflow_kernel<<<BATCH / TILE_B, 256, 0, stream>>>(nptr, WhT, out);
}
